// MODEL_43181601194154
// MI455X (gfx1250) — matrix operand order repaired and checked by reading, not compiled, not run on hardware
//
#include <hip/hip_runtime.h>
#include <hip/hip_bf16.h>
#include <cmath>

// ---------------------------------------------------------------------------
// DKVMN forward for MI455X (gfx1250, wave32, WMMA).
//   B=256, T=500, DK=128, DV=256, M=50, DF=DV+DK=384, BT=128000
// Strategy:
//   1) gather embeddings -> bf16 activations
//   2) E=sigmoid(qa@We.T+be), A=tanh(qa@Wa.T+ba) as bf16-WMMA GEMMs (hoisted
//      out of the recurrence; they don't depend on Mv)
//   3) attention weights w = softmax(q@Mk.T) (small, VALU)
//   4) sequential scan: 1 workgroup per batch; thread d owns column d of Mv,
//      which lives in 50 REGISTERS per thread (no LDS round-trips on the
//      serial path); w broadcast via double-buffered LDS, one barrier/step;
//      next-step E/A/w prefetched to hide global latency in the serial chain
//   5) h = tanh(feat@Wr.T+br) via bf16 WMMA
//   6) pred/probs/masked-BCE + reduction, finalize loss
// ---------------------------------------------------------------------------

typedef __attribute__((ext_vector_type(16))) __bf16 v16bf;
typedef __attribute__((ext_vector_type(8)))  __bf16 v8bf;
typedef __attribute__((ext_vector_type(8)))  float  v8f;

#define BT_  128000
#define B_   256
#define T_   500
#define DK_  128
#define DV_  256
#define M_   50
#define DF_  384

__device__ inline v8f zero_v8f() {
  v8f z = {0.f,0.f,0.f,0.f,0.f,0.f,0.f,0.f};
  return z;
}

// Build a 16-bit WMMA A-fragment (16x32 MxK) from row-major bf16 src.
// lane<16:  M=lane,    K = 0..7  and 16..23
// lane>=16: M=lane-16, K = 8..15 and 24..31
__device__ inline v16bf load_a_frag(const __bf16* base, long row, int ld,
                                    int k0, int lhalf) {
  const __bf16* ap = base + row * (long)ld + k0 + lhalf * 8;
  v8bf lo = *(const v8bf*)ap;
  v8bf hi = *(const v8bf*)(ap + 16);
  v16bf a;
#pragma unroll
  for (int i = 0; i < 8; ++i) { a[i] = lo[i]; a[8 + i] = hi[i]; }
  return a;
}

// B-fragment (32x16 KxN) for X @ W.T:  B[k][col] = W[col][k].
// lanes 0-15: K=k0+0..7 and k0+16..23 (col=lane), lanes 16-31: K=k0+8..15 and k0+24..31.
__device__ inline v16bf load_b_frag(const __bf16* W, int col, int ld,
                                    int k0, int lhalf) {
  const __bf16* bp = W + (long)col * ld + k0 + lhalf * 8;
  v8bf lo = *(const v8bf*)bp;
  v8bf hi = *(const v8bf*)(bp + 16);
  v16bf b;
#pragma unroll
  for (int i = 0; i < 8; ++i) { b[i] = lo[i]; b[8 + i] = hi[i]; }
  return b;
}

// --------------------------- weight conversion -----------------------------
__global__ void wconv_kernel(const float* We, const float* Wa, const float* Wr,
                             __bf16* Web, __bf16* Wab, __bf16* Wrb) {
  int i = blockIdx.x * 256 + threadIdx.x;
  if (i < DV_ * DV_) {
    Web[i] = (__bf16)We[i];
    Wab[i] = (__bf16)Wa[i];
  }
  if (i < DK_ * DF_) Wrb[i] = (__bf16)Wr[i];
}

// --------------------------- embedding gather ------------------------------
__global__ void gather_kernel(const int* qd, const int* qad,
                              const float* qtab, const float* qatab,
                              __bf16* qa_bf, __bf16* feat) {
  long i = (long)blockIdx.x * blockDim.x + threadIdx.x;
  long stride = (long)gridDim.x * blockDim.x;
  long totQA = (long)BT_ * DV_;
  for (long idx = i; idx < totQA; idx += stride) {
    long n = idx / DV_;
    int  d = (int)(idx - n * DV_);
    qa_bf[idx] = (__bf16)qatab[(long)qad[n] * DV_ + d];
  }
  long totQ = (long)BT_ * DK_;
  for (long idx = i; idx < totQ; idx += stride) {
    long n = idx / DK_;
    int  k = (int)(idx - n * DK_);
    feat[n * DF_ + DV_ + k] = (__bf16)qtab[(long)qd[n] * DK_ + k];
  }
}

// ------------------- fused erase/add GEMMs (bf16 WMMA) ---------------------
// 1 block = 4 waves; block owns one 16-row M tile; wave w covers N cols
// [64w, 64w+64). 8 K-steps of 32. Fuses E and A (same A-fragment).
__global__ void ea_gemm_kernel(const __bf16* qa, const __bf16* Web,
                               const __bf16* Wab, const float* be,
                               const float* ba, __bf16* E, __bf16* A) {
  const int mtile = blockIdx.x;
  const int wave  = threadIdx.x >> 5;
  const int lane  = threadIdx.x & 31;
  const int lhalf = lane >> 4;
  const int lidx  = lane & 15;
  const long arow = (long)mtile * 16 + lidx;

  v8f accE[4], accA[4];
#pragma unroll
  for (int t = 0; t < 4; ++t) { accE[t] = zero_v8f(); accA[t] = zero_v8f(); }

  for (int kt = 0; kt < 8; ++kt) {
    const int k0 = kt * 32;
    v16bf afr = load_a_frag(qa, arow, DV_, k0, lhalf);
#pragma unroll
    for (int nt = 0; nt < 4; ++nt) {
      const int col = (wave * 4 + nt) * 16 + lidx;
      v16bf bE = load_b_frag(Web, col, DV_, k0, lhalf);
      accE[nt] = __builtin_amdgcn_wmma_f32_16x16x32_bf16(
          false, afr, false, bE, (short)0, accE[nt], false, false);
      v16bf bA = load_b_frag(Wab, col, DV_, k0, lhalf);
      accA[nt] = __builtin_amdgcn_wmma_f32_16x16x32_bf16(
          false, afr, false, bA, (short)0, accA[nt], false, false);
    }
  }

#pragma unroll
  for (int nt = 0; nt < 4; ++nt) {
    const int col = (wave * 4 + nt) * 16 + lidx;
    const float bev = be[col], bav = ba[col];
#pragma unroll
    for (int r = 0; r < 8; ++r) {
      const long row = (long)mtile * 16 + r + lhalf * 8;
      float ev = 1.f / (1.f + __expf(-(accE[nt][r] + bev)));
      float av = tanhf(accA[nt][r] + bav);
      E[row * DV_ + col] = (__bf16)ev;
      A[row * DV_ + col] = (__bf16)av;
    }
  }
}

// ------------------- attention logits + softmax (M=50) ---------------------
// block of 256 threads handles 8 rows; q rows staged in LDS.
__global__ void attn_kernel(const __bf16* feat, const float* Mk, float* w) {
  __shared__ float q_s[8][DK_];
  __shared__ float lg[8][M_];
  const int n0 = blockIdx.x * 8;
  for (int i = threadIdx.x; i < 8 * DK_; i += 256) {
    int nl = i / DK_, k = i % DK_;
    q_s[nl][k] = (float)feat[(long)(n0 + nl) * DF_ + DV_ + k];
  }
  __syncthreads();
  for (int i = threadIdx.x; i < 8 * M_; i += 256) {
    int nl = i / M_, m = i % M_;
    const float* mk = Mk + m * DK_;
    float s = 0.f;
#pragma unroll 8
    for (int k = 0; k < DK_; ++k) s += q_s[nl][k] * mk[k];
    lg[nl][m] = s;
  }
  __syncthreads();
  if (threadIdx.x < 8) {
    const int nl = threadIdx.x;
    float mx = lg[nl][0];
    for (int m = 1; m < M_; ++m) mx = fmaxf(mx, lg[nl][m]);
    float sum = 0.f;
    for (int m = 0; m < M_; ++m) { lg[nl][m] = __expf(lg[nl][m] - mx); sum += lg[nl][m]; }
    const float inv = 1.f / sum;
    float* wp = w + (long)(n0 + nl) * M_;
    for (int m = 0; m < M_; ++m) wp[m] = lg[nl][m] * inv;
  }
}

// --------------------------- recurrent scan --------------------------------
// 1 block per batch row. Thread d owns Mv[:, d] in 50 REGISTERS (fully
// unrolled, constant indices -> SROA to VGPRs). Softmax weights broadcast via
// double-buffered LDS: one barrier per step. Write-gate applied as a
// multiplicative factor (block-uniform) to keep a single unrolled loop body.
__global__ void scan_kernel(const int* qd, const float* Mv0,
                            const __bf16* E, const __bf16* A,
                            const float* w, __bf16* feat) {
  __shared__ float w_s[2][M_];
  const int b = blockIdx.x;
  const int d = threadIdx.x;

  float mv[M_];
#pragma unroll
  for (int m = 0; m < M_; ++m) mv[m] = Mv0[m * DV_ + d];

  for (int t = 0; t < T_; ++t) {
    const long n   = (long)b * T_ + t;
    const int  buf = t & 1;
    if (d < M_) w_s[buf][d] = w[n * M_ + d];

    const float ev  = (float)E[n * DV_ + d];
    const float av  = (float)A[n * DV_ + d];
    const float wrf = (qd[n] >= 1) ? 1.f : 0.f;   // block-uniform write gate
    if (t + 1 < T_) {                              // hide latency of the
      __builtin_prefetch(&E[(n + 1) * DV_ + d], 0, 3);  // serial chain
      __builtin_prefetch(&A[(n + 1) * DV_ + d], 0, 3);
      __builtin_prefetch(&w[(n + 1) * M_], 0, 3);
    }
    __syncthreads();

    float r = 0.f;
#pragma unroll
    for (int m = 0; m < M_; ++m) {
      const float wm  = w_s[buf][m];
      const float wmu = wm * wrf;
      r += wm * mv[m];
      mv[m] = mv[m] * (1.f - wmu * ev) + wmu * av;
    }
    feat[n * DF_ + d] = (__bf16)r;
  }
}

// ------------------- head GEMM: h = tanh(feat @ Wr.T + br) -----------------
// 1 block = 4 waves; block owns one 16-row M tile; wave w covers N [32w,32w+32).
__global__ void head_gemm_kernel(const __bf16* feat, const __bf16* Wrb,
                                 const float* br, __bf16* h) {
  const int mtile = blockIdx.x;
  const int wave  = threadIdx.x >> 5;
  const int lane  = threadIdx.x & 31;
  const int lhalf = lane >> 4;
  const int lidx  = lane & 15;
  const long arow = (long)mtile * 16 + lidx;

  v8f acc[2];
  acc[0] = zero_v8f(); acc[1] = zero_v8f();

  for (int kt = 0; kt < 12; ++kt) {
    const int k0 = kt * 32;
    v16bf afr = load_a_frag(feat, arow, DF_, k0, lhalf);
#pragma unroll
    for (int nt = 0; nt < 2; ++nt) {
      const int col = wave * 32 + nt * 16 + lidx;
      v16bf bfr = load_b_frag(Wrb, col, DF_, k0, lhalf);
      acc[nt] = __builtin_amdgcn_wmma_f32_16x16x32_bf16(
          false, afr, false, bfr, (short)0, acc[nt], false, false);
    }
  }

#pragma unroll
  for (int nt = 0; nt < 2; ++nt) {
    const int col = wave * 32 + nt * 16 + lidx;
    const float bv = br[col];
#pragma unroll
    for (int r = 0; r < 8; ++r) {
      const long row = (long)mtile * 16 + r + lhalf * 8;
      h[row * DK_ + col] = (__bf16)tanhf(acc[nt][r] + bv);
    }
  }
}

// ------------------- pred, probs, masked BCE reduction ---------------------
__global__ void pred_loss_kernel(const __bf16* h, const float* Wp,
                                 const float* bp, const float* target,
                                 float* out, float* accum) {
  const long n = (long)blockIdx.x * 256 + threadIdx.x;
  const __bf16* hp = h + n * DK_;
  float s = 0.f;
#pragma unroll 8
  for (int k = 0; k < DK_; ++k) s += (float)hp[k] * Wp[k];
  const float pred = s + bp[0];
  const float tgt  = target[n];
  out[1 + n] = 1.f / (1.f + __expf(-pred));
  const float per  = fmaxf(pred, 0.f) - pred * tgt + log1pf(__expf(-fabsf(pred)));
  const bool  msk  = tgt >= 0.f;

  __shared__ float s_sum[256];
  __shared__ float s_cnt[256];
  s_sum[threadIdx.x] = msk ? per : 0.f;
  s_cnt[threadIdx.x] = msk ? 1.f : 0.f;
  __syncthreads();
  for (int st = 128; st > 0; st >>= 1) {
    if (threadIdx.x < st) {
      s_sum[threadIdx.x] += s_sum[threadIdx.x + st];
      s_cnt[threadIdx.x] += s_cnt[threadIdx.x + st];
    }
    __syncthreads();
  }
  if (threadIdx.x == 0) {
    atomicAdd(&accum[0], s_sum[0]);
    atomicAdd(&accum[1], s_cnt[0]);
  }
}

__global__ void finalize_kernel(const float* accum, float* out) {
  out[0] = accum[0] / fmaxf(accum[1], 1.f);
}

// ---------------------------------------------------------------------------
extern "C" void kernel_launch(void* const* d_in, const int* in_sizes, int n_in,
                              void* d_out, int out_size, void* d_ws,
                              size_t ws_size, hipStream_t stream) {
  const int*   qd     = (const int*)  d_in[0];
  const int*   qad    = (const int*)  d_in[1];
  const float* target = (const float*)d_in[2];
  const float* qtab   = (const float*)d_in[3];
  const float* qatab  = (const float*)d_in[4];
  const float* Mk     = (const float*)d_in[5];
  const float* Mv0    = (const float*)d_in[6];
  const float* We     = (const float*)d_in[7];
  const float* be     = (const float*)d_in[8];
  const float* Wa     = (const float*)d_in[9];
  const float* ba     = (const float*)d_in[10];
  const float* Wr     = (const float*)d_in[11];
  const float* br     = (const float*)d_in[12];
  const float* Wp     = (const float*)d_in[13];
  const float* bp     = (const float*)d_in[14];
  float* out = (float*)d_out;

  // workspace carve-up (256B aligned)
  char* ws = (char*)d_ws;
  size_t off = 0;
  auto carve = [&](size_t bytes) -> char* {
    char* p = ws + off;
    off = (off + bytes + 255) & ~(size_t)255;
    return p;
  };
  __bf16* qa_bf = (__bf16*)carve((size_t)BT_ * DV_ * 2);
  __bf16* feat  = (__bf16*)carve((size_t)BT_ * DF_ * 2);
  __bf16* Ebuf  = (__bf16*)carve((size_t)BT_ * DV_ * 2);
  __bf16* Abuf  = (__bf16*)carve((size_t)BT_ * DV_ * 2);
  float*  wbuf  = (float*) carve((size_t)BT_ * M_  * 4);
  __bf16* hbuf  = (__bf16*)carve((size_t)BT_ * DK_ * 2);
  __bf16* Web   = (__bf16*)carve((size_t)DV_ * DV_ * 2);
  __bf16* Wab   = (__bf16*)carve((size_t)DV_ * DV_ * 2);
  __bf16* Wrb   = (__bf16*)carve((size_t)DK_ * DF_ * 2);
  float*  accum = (float*) carve(2 * 4);

  hipMemsetAsync(accum, 0, 2 * sizeof(float), stream);

  wconv_kernel<<<(DV_ * DV_ + 255) / 256, 256, 0, stream>>>(We, Wa, Wr,
                                                            Web, Wab, Wrb);
  gather_kernel<<<2048, 256, 0, stream>>>(qd, qad, qtab, qatab, qa_bf, feat);
  ea_gemm_kernel<<<BT_ / 16, 128, 0, stream>>>(qa_bf, Web, Wab, be, ba,
                                               Ebuf, Abuf);
  attn_kernel<<<BT_ / 8, 256, 0, stream>>>(feat, Mk, wbuf);
  scan_kernel<<<B_, 256, 0, stream>>>(qd, Mv0, Ebuf, Abuf, wbuf, feat);
  head_gemm_kernel<<<BT_ / 16, 128, 0, stream>>>(feat, Wrb, br, hbuf);
  pred_loss_kernel<<<BT_ / 256, 256, 0, stream>>>(hbuf, Wp, bp, target,
                                                  out, accum);
  finalize_kernel<<<1, 1, 0, stream>>>(accum, out);
}
